// MultiHeadGraphAttention_60825326846769
// MI455X (gfx1250) — matrix operand order repaired and checked by reading, not compiled, not run on hardware
//
#include <hip/hip_runtime.h>

#define BB 8
#define NN 1024
#define IND 256
#define OUTD 256
#define HH 4
#define HD 64

typedef __attribute__((ext_vector_type(16))) __bf16 v16bf;
typedef __attribute__((ext_vector_type(8)))  float  v8f;

union ABfrag { uint4 u[2]; v16bf v; };

__device__ __forceinline__ unsigned short f2bf(float f) {
  unsigned int u = __float_as_uint(f);
  unsigned int r = u + 0x7FFFu + ((u >> 16) & 1u);   // round-to-nearest-even
  return (unsigned short)(r >> 16);
}

// A-fragment (16x32 bf16, row-major source, leading dim ld elements).
// Lane l (0..15): row=l, K groups {0..7, 16..23}; lane l+16: row=l, K {8..15, 24..31}.
__device__ __forceinline__ v16bf load_a_frag(const unsigned short* base, int ld, int lane) {
  const unsigned short* p = base + (size_t)(lane & 15) * ld + ((lane >> 4) << 3);
  ABfrag f;
  f.u[0] = *(const uint4*)(p);
  f.u[1] = *(const uint4*)(p + 16);
  return f.v;
}

// B-fragment (32x16 bf16, COLUMN-major source: baseT + col*ld + k contiguous in k).
// Lane l (0..15): col=l, K groups {0..7, 16..23}; lane l+16: col=l, K {8..15, 24..31}.
__device__ __forceinline__ v16bf load_b_frag(const unsigned short* baseT, int ld, int lane) {
  const unsigned short* p = baseT + (size_t)(lane & 15) * ld + ((lane >> 4) << 3);
  ABfrag f;
  f.u[0] = *(const uint4*)(p);
  f.u[1] = *(const uint4*)(p + 16);
  return f.v;
}

__device__ __forceinline__ v8f wmma_bf16(v16bf a, v16bf b, v8f c) {
  return __builtin_amdgcn_wmma_f32_16x16x32_bf16(false, a, false, b, (short)0, c, false, false);
}

// ---------------- conversions ----------------
__global__ __launch_bounds__(256) void k_cvt_h(const float* __restrict__ h,
                                               unsigned short* __restrict__ hb, int n) {
  int i = blockIdx.x * 256 + threadIdx.x;
  if (i < n) hb[i] = f2bf(h[i]);
}

__global__ __launch_bounds__(256) void k_cvt_w(const float* __restrict__ W,
                                               const float* __restrict__ pw,
                                               unsigned short* __restrict__ wT,
                                               unsigned short* __restrict__ pwb) {
  int i = blockIdx.x * 256 + threadIdx.x;
  if (i < HH * IND * HD) {              // W [H][IN][HD] -> wT [H][HD][IN]
    int hh = i / (IND * HD);
    int rem = i % (IND * HD);
    int k = rem / HD, d = rem % HD;
    wT[((size_t)hh * HD + d) * IND + k] = f2bf(W[i]);
  }
  if (i < OUTD * OUTD) pwb[i] = f2bf(pw[i]);   // proj_w [o][d] is already B-col-major
}

// ------- K1: Wh = h x W -> Wh^T bf16 [B][H][HD][N]; fused si/sj scores -------
__global__ __launch_bounds__(256) void k_gemm_wh(const unsigned short* __restrict__ hb,
                                                 const unsigned short* __restrict__ wT,
                                                 const float* __restrict__ a1,
                                                 const float* __restrict__ a2,
                                                 unsigned short* __restrict__ whT,
                                                 float* __restrict__ si,
                                                 float* __restrict__ sj) {
  extern __shared__ uint4 smem4[];
  unsigned short* lds = (unsigned short*)smem4;          // W head slice [HD][IND] = 32 KB
  int b = blockIdx.z, hh = blockIdx.y;
  int wave = threadIdx.x >> 5, lane = threadIdx.x & 31;
  int m0 = blockIdx.x * 128 + wave * 16;

  {  // cooperative stage of B into LDS (2048 uint4)
    const uint4* src = (const uint4*)(wT + (size_t)hh * HD * IND);
    uint4* dst = (uint4*)lds;
    for (int i = threadIdx.x; i < (HD * IND) / 8; i += 256) dst[i] = src[i];
  }
  __syncthreads();

  const unsigned short* A = hb + (size_t)b * NN * IND;
  v8f acc[4] = {};
  for (int k = 0; k < IND; k += 32) {
    v16bf a = load_a_frag(A + (size_t)m0 * IND + k, IND, lane);
#pragma unroll
    for (int nt = 0; nt < 4; ++nt) {
      v16bf bf = load_b_frag(lds + (size_t)(nt * 16) * IND + k, IND, lane);
      acc[nt] = wmma_bf16(a, bf, acc[nt]);
    }
  }

  int hl = lane >> 4, cl = lane & 15;
  unsigned short* O = whT + (size_t)(b * HH + hh) * HD * NN;
#pragma unroll
  for (int nt = 0; nt < 4; ++nt) {
    int d = nt * 16 + cl;
    unsigned int w0 = (unsigned)f2bf(acc[nt][0]) | ((unsigned)f2bf(acc[nt][1]) << 16);
    unsigned int w1 = (unsigned)f2bf(acc[nt][2]) | ((unsigned)f2bf(acc[nt][3]) << 16);
    unsigned int w2 = (unsigned)f2bf(acc[nt][4]) | ((unsigned)f2bf(acc[nt][5]) << 16);
    unsigned int w3 = (unsigned)f2bf(acc[nt][6]) | ((unsigned)f2bf(acc[nt][7]) << 16);
    uint4 pk; pk.x = w0; pk.y = w1; pk.z = w2; pk.w = w3;
    *(uint4*)(O + (size_t)d * NN + m0 + hl * 8) = pk;   // 8 consecutive rows m
  }

  // fused attention scores: si[m] = Wh[m,:].a1 ; sj[m] = Wh[m,:].a2 (f32 accumulators)
  float a1v[4], a2v[4];
#pragma unroll
  for (int nt = 0; nt < 4; ++nt) {
    a1v[nt] = a1[hh * HD + nt * 16 + cl];
    a2v[nt] = a2[hh * HD + nt * 16 + cl];
  }
  size_t sbase = (size_t)(b * HH + hh) * NN;
#pragma unroll
  for (int r = 0; r < 8; ++r) {
    float p1 = 0.f, p2 = 0.f;
#pragma unroll
    for (int nt = 0; nt < 4; ++nt) { p1 += acc[nt][r] * a1v[nt]; p2 += acc[nt][r] * a2v[nt]; }
#pragma unroll
    for (int off = 8; off; off >>= 1) {       // reduce across the 16 lanes of a half
      p1 += __shfl_xor(p1, off, 32);
      p2 += __shfl_xor(p2, off, 32);
    }
    if (cl == 0) {
      int m = m0 + hl * 8 + r;
      si[sbase + m] = p1;
      sj[sbase + m] = p2;
    }
  }
}

// -------- K2: masked leaky-relu softmax -> alpha bf16 [B][H][N][N] --------
__global__ __launch_bounds__(256) void k_softmax(const float* __restrict__ si,
                                                 const float* __restrict__ sj,
                                                 const int* __restrict__ adj,
                                                 unsigned short* __restrict__ alpha) {
  int b = blockIdx.z, hh = blockIdx.y;
  int wave = threadIdx.x >> 5, lane = threadIdx.x & 31;
  int row = blockIdx.x * 8 + wave;
  int bh = b * HH + hh;
  float s_i = si[bh * NN + row];
  const float* sjp = sj + (size_t)bh * NN;
  const int* arow = adj + (size_t)row * NN;
  float ev[32];
  float vmax = -1e30f;
#pragma unroll
  for (int it = 0; it < 32; ++it) {
    int m = it * 32 + lane;
    float e = s_i + sjp[m];
    e = e > 0.f ? e : 0.2f * e;           // leaky_relu(0.2)
    e = arow[m] != 0 ? e : -1e9f;         // mask
    ev[it] = e;
    vmax = fmaxf(vmax, e);
  }
  for (int off = 16; off; off >>= 1) vmax = fmaxf(vmax, __shfl_xor(vmax, off, 32));
  float sum = 0.f;
#pragma unroll
  for (int it = 0; it < 32; ++it) { float p = __expf(ev[it] - vmax); ev[it] = p; sum += p; }
  for (int off = 16; off; off >>= 1) sum += __shfl_xor(sum, off, 32);
  float inv = __frcp_rn(sum);             // self-loops guarantee sum > 0
  unsigned short* out = alpha + ((size_t)bh * NN + row) * NN;
#pragma unroll
  for (int it = 0; it < 32; ++it) out[it * 32 + lane] = f2bf(ev[it] * inv);
}

// -------- K3: out = alpha x Wh -> h_multi bf16 [B][N][OUT] (heads concat) --------
__global__ __launch_bounds__(256) void k_attn_out(const unsigned short* __restrict__ alpha,
                                                  const unsigned short* __restrict__ whT,
                                                  unsigned short* __restrict__ hmulti) {
  extern __shared__ uint4 smem4[];
  unsigned short* lds = (unsigned short*)smem4;          // Wh^T slice [HD][N] = 128 KB
  int b = blockIdx.z, hh = blockIdx.y;
  int wave = threadIdx.x >> 5, lane = threadIdx.x & 31;
  int m0 = blockIdx.x * 128 + wave * 16;
  int bh = b * HH + hh;

  {  // cooperative stage of B into LDS (8192 uint4)
    const uint4* src = (const uint4*)(whT + (size_t)bh * HD * NN);
    uint4* dst = (uint4*)lds;
    for (int i = threadIdx.x; i < (HD * NN) / 8; i += 256) dst[i] = src[i];
  }
  __syncthreads();

  const unsigned short* A = alpha + (size_t)bh * NN * NN;   // [N][N] row-major
  v8f acc[4] = {};
  for (int k = 0; k < NN; k += 32) {
    __builtin_prefetch(A + (size_t)m0 * NN + k + 128, 0, 1);
    v16bf a = load_a_frag(A + (size_t)m0 * NN + k, NN, lane);
#pragma unroll
    for (int nt = 0; nt < 4; ++nt) {
      v16bf bf = load_b_frag(lds + (size_t)(nt * 16) * NN + k, NN, lane);
      acc[nt] = wmma_bf16(a, bf, acc[nt]);
    }
  }
  unsigned short* O = hmulti + (size_t)b * NN * OUTD;
  int hl = lane >> 4;
#pragma unroll
  for (int nt = 0; nt < 4; ++nt) {
    int col = hh * HD + nt * 16 + (lane & 15);
#pragma unroll
    for (int r = 0; r < 8; ++r) {
      int m = m0 + hl * 8 + r;
      O[(size_t)m * OUTD + col] = f2bf(acc[nt][r]);
    }
  }
}

// -------- K4: projection GEMM + bias + residual + LayerNorm -> d_out fp32 --------
__global__ __launch_bounds__(256) void k_proj_ln(const unsigned short* __restrict__ hmulti,
                                                 const unsigned short* __restrict__ pwb,
                                                 const float* __restrict__ h0,
                                                 const float* __restrict__ pb,
                                                 const float* __restrict__ gamma,
                                                 const float* __restrict__ beta,
                                                 float* __restrict__ out) {
  extern __shared__ uint4 smem4[];
  unsigned short* lds = (unsigned short*)smem4;          // proj_w bf16 [O][D] = 128 KB
  int b = blockIdx.y;
  int wave = threadIdx.x >> 5, lane = threadIdx.x & 31;
  int m0 = blockIdx.x * 128 + wave * 16;

  {  // cooperative stage of B into LDS (8192 uint4)
    const uint4* src = (const uint4*)pwb;
    uint4* dst = (uint4*)lds;
    for (int i = threadIdx.x; i < (OUTD * OUTD) / 8; i += 256) dst[i] = src[i];
  }
  __syncthreads();

  const unsigned short* A = hmulti + (size_t)b * NN * OUTD;
  v8f acc[16] = {};
  for (int k = 0; k < OUTD; k += 32) {
    v16bf a = load_a_frag(A + (size_t)m0 * OUTD + k, OUTD, lane);
#pragma unroll
    for (int nt = 0; nt < 16; ++nt) {
      v16bf bf = load_b_frag(lds + (size_t)(nt * 16) * OUTD + k, OUTD, lane);
      acc[nt] = wmma_bf16(a, bf, acc[nt]);
    }
  }

  // fused epilogue: +bias, +residual, LayerNorm per row (16 lanes of a half hold one row)
  int hl = lane >> 4, cl = lane & 15;
#pragma unroll
  for (int r = 0; r < 8; ++r) {
    int m = m0 + hl * 8 + r;
    const float* h0r = h0 + ((size_t)b * NN + m) * OUTD;
    float vals[16];
    float s = 0.f, s2 = 0.f;
#pragma unroll
    for (int nt = 0; nt < 16; ++nt) {
      int col = nt * 16 + cl;
      float v = acc[nt][r] + pb[col] + h0r[col];
      vals[nt] = v; s += v; s2 += v * v;
    }
#pragma unroll
    for (int off = 8; off; off >>= 1) {
      s  += __shfl_xor(s,  off, 32);
      s2 += __shfl_xor(s2, off, 32);
    }
    float mu  = s * (1.f / OUTD);
    float var = s2 * (1.f / OUTD) - mu * mu;
    float rstd = rsqrtf(var + 1e-5f);
    float* orow = out + ((size_t)b * NN + m) * OUTD;
#pragma unroll
    for (int nt = 0; nt < 16; ++nt) {
      int col = nt * 16 + cl;
      orow[col] = (vals[nt] - mu) * rstd * gamma[col] + beta[col];
    }
  }
}

extern "C" void kernel_launch(void* const* d_in, const int* in_sizes, int n_in,
                              void* d_out, int out_size, void* d_ws, size_t ws_size,
                              hipStream_t stream) {
  const float* h     = (const float*)d_in[0];
  const int*   adj   = (const int*)d_in[1];
  const float* W     = (const float*)d_in[2];
  const float* a1    = (const float*)d_in[3];
  const float* a2    = (const float*)d_in[4];
  const float* pw    = (const float*)d_in[5];
  const float* pbias = (const float*)d_in[6];
  const float* gamma = (const float*)d_in[7];
  const float* beta  = (const float*)d_in[8];
  float* out = (float*)d_out;

  char* ws = (char*)d_ws;
  unsigned short* hb    = (unsigned short*)(ws);                                       // 4 MB
  unsigned short* wT    = (unsigned short*)(ws + (4u << 20));                          // 128 KB
  unsigned short* pwb   = (unsigned short*)(ws + (4u << 20) + (128u << 10));           // 128 KB
  unsigned short* whT   = (unsigned short*)(ws + (4u << 20) + (256u << 10));           // 4 MB
  float*          si    = (float*)(ws + (8u << 20) + (256u << 10));                    // 128 KB
  float*          sj    = (float*)(ws + (8u << 20) + (384u << 10));                    // 128 KB
  unsigned short* alpha = (unsigned short*)(ws + (8u << 20) + (512u << 10));           // 64 MB
  unsigned short* hm    = (unsigned short*)(ws + (72u << 20) + (512u << 10));          // 4 MB

  // conversions
  k_cvt_h<<<(BB * NN * IND) / 256, 256, 0, stream>>>(h, hb, BB * NN * IND);
  k_cvt_w<<<(OUTD * OUTD) / 256, 256, 0, stream>>>(W, pw, wT, pwb);

  // Wh^T + fused si/sj   (LDS: 32 KB)
  k_gemm_wh<<<dim3(NN / 128, HH, BB), 256, HD * IND * 2, stream>>>(hb, wT, a1, a2, whT, si, sj);

  // masked softmax -> alpha
  k_softmax<<<dim3(NN / 8, HH, BB), 256, 0, stream>>>(si, sj, adj, alpha);

  // attention output -> h_multi   (LDS: 128 KB)
  k_attn_out<<<dim3(NN / 128, HH, BB), 256, HD * NN * 2, stream>>>(alpha, whT, hm);

  // projection + bias + residual + layernorm   (LDS: 128 KB)
  k_proj_ln<<<dim3(NN / 128, BB), 256, OUTD * OUTD * 2, stream>>>(hm, pwb, h, pbias, gamma, beta, out);
}
